// Prefill_40054865002721
// MI455X (gfx1250) — matrix operand order repaired and checked by reading, not compiled, not run on hardware
//
#include <hip/hip_runtime.h>
#include <math.h>

// ---------------- model constants ----------------
#define BB 4
#define SS 1024
#define EE 768
#define HH 12
#define LL 12
#define VV 50257
#define FF 3072
#define DH 64
#define E3 (3*EE)

// ---------------- WMMA types ----------------
typedef __attribute__((ext_vector_type(16))) __bf16 v16bf;
typedef __attribute__((ext_vector_type(8)))  __bf16 v8bf;
typedef __attribute__((ext_vector_type(4)))  __bf16 v4bf;
typedef __attribute__((ext_vector_type(8)))  float  v8f;

__device__ __forceinline__ v8f vzero8() {
  v8f v;
#pragma unroll
  for (int i = 0; i < 8; ++i) v[i] = 0.0f;
  return v;
}

// A fragment: 16x32 bf16 tile, row-major in LDS (stride ld elems).
// lane 0-15: row=lane, K 0-7 & 16-23 ; lane 16-31: row=lane-16, K 8-15 & 24-31.
__device__ __forceinline__ v16bf frag_a(const __bf16* p, int ld, int lane) {
  const int row = lane & 15, half = lane >> 4;
  union { v16bf v; v8bf h[2]; } u;
  u.h[0] = *reinterpret_cast<const v8bf*>(p + row * ld + half * 8);
  u.h[1] = *reinterpret_cast<const v8bf*>(p + row * ld + 16 + half * 8);
  return u.v;
}
// B fragment: 32x16 bf16 tile stored contraction-major per column: LDS layout
// [n][k] (stride ld). lane n holds column n, K = half*8..half*8+7 & 16+half*8..16+half*8+7.
__device__ __forceinline__ v16bf frag_b(const __bf16* p, int ld, int lane) {
  const int n = lane & 15, half = lane >> 4;
  union { v16bf v; v8bf h[2]; } u;
  u.h[0] = *reinterpret_cast<const v8bf*>(p + n * ld + half * 8);
  u.h[1] = *reinterpret_cast<const v8bf*>(p + n * ld + 16 + half * 8);
  return u.v;
}

__device__ __forceinline__ v8f wmma_bf16(v16bf a, v16bf b, v8f c) {
  return __builtin_amdgcn_wmma_f32_16x16x32_bf16(false, a, false, b, (short)0, c,
                                                 false, false);
}

// ---------------- f32 [K,N] -> bf16 transposed [N,K] weight prep ----------------
// One-shot per launch; gives GEMM a contiguous B-staging path (b128 LDS stores).
__global__ __launch_bounds__(256) void cvtT_bf16_kernel(const float* __restrict__ src,
                                                        __bf16* __restrict__ dst,
                                                        int K, int N) {
  const size_t slab = (size_t)K * N;
  const float* s = src + blockIdx.y * slab;
  __bf16* d = dst + blockIdx.y * slab;
  const size_t i = ((size_t)blockIdx.x * 256 + threadIdx.x) * 4;  // output idx
  if (i >= slab) return;
  const int n = (int)(i / K);
  const int k = (int)(i % K);
  v4bf pk = { (__bf16)s[(size_t)(k + 0) * N + n], (__bf16)s[(size_t)(k + 1) * N + n],
              (__bf16)s[(size_t)(k + 2) * N + n], (__bf16)s[(size_t)(k + 3) * N + n] };
  *reinterpret_cast<v4bf*>(&d[i]) = pk;
}

// ---------------- embedding ----------------
__global__ __launch_bounds__(192) void embed_kernel(const int* __restrict__ ids,
                                                    const float* __restrict__ wte,
                                                    const float* __restrict__ wpe,
                                                    float* __restrict__ x) {
  const int bs = blockIdx.x;          // [0, B*S)
  const int s = bs & (SS - 1);
  const int id = ids[bs];
  const int e = threadIdx.x * 4;
  const float4 a = *reinterpret_cast<const float4*>(wte + (size_t)id * EE + e);
  const float4 p = *reinterpret_cast<const float4*>(wpe + (size_t)s * EE + e);
  float4 r;
  r.x = a.x + p.x; r.y = a.y + p.y; r.z = a.z + p.z; r.w = a.w + p.w;
  *reinterpret_cast<float4*>(x + (size_t)bs * EE + e) = r;
}

// ---------------- block reductions (256 threads) ----------------
__device__ __forceinline__ float block_sum256(float v, float* red) {
  const int lane = threadIdx.x & 31, wave = threadIdx.x >> 5;
#pragma unroll
  for (int d = 16; d > 0; d >>= 1) v += __shfl_xor(v, d, 32);
  if (lane == 0) red[wave] = v;
  __syncthreads();
  float t = 0.0f;
#pragma unroll
  for (int i = 0; i < 8; ++i) t += red[i];
  __syncthreads();
  return t;
}
__device__ __forceinline__ float block_max256(float v, float* red) {
  const int lane = threadIdx.x & 31, wave = threadIdx.x >> 5;
#pragma unroll
  for (int d = 16; d > 0; d >>= 1) v = fmaxf(v, __shfl_xor(v, d, 32));
  if (lane == 0) red[wave] = v;
  __syncthreads();
  float t = -3.0e38f;
#pragma unroll
  for (int i = 0; i < 8; ++i) t = fmaxf(t, red[i]);
  __syncthreads();
  return t;
}

// ---------------- layernorm (one row per block) ----------------
__global__ __launch_bounds__(256) void ln_kernel(const float* __restrict__ x,
                                                 const float* __restrict__ w,
                                                 const float* __restrict__ b,
                                                 float* __restrict__ y) {
  __shared__ float red[8];
  const size_t row = blockIdx.x;
  const float* xr = x + row * EE;
  float v0 = xr[threadIdx.x];
  float v1 = xr[threadIdx.x + 256];
  float v2 = xr[threadIdx.x + 512];
  const float s = block_sum256(v0 + v1 + v2, red);
  const float mu = s * (1.0f / EE);
  const float d0 = v0 - mu, d1 = v1 - mu, d2 = v2 - mu;
  const float q = block_sum256(d0 * d0 + d1 * d1 + d2 * d2, red);
  const float rinv = rsqrtf(q * (1.0f / EE) + 1e-5f);
  float* yr = y + row * EE;
  int e = threadIdx.x;
  yr[e] = d0 * rinv * w[e] + b[e];
  e += 256; yr[e] = d1 * rinv * w[e] + b[e];
  e += 256; yr[e] = d2 * rinv * w[e] + b[e];
}

// ---------------- generic bf16 WMMA GEMM ----------------
// out[M,N] = act( A[M,K] @ W[K,N] + bias[N] + (res ? res[M,N] : 0) )
// W supplied pre-transposed+bf16: Wt[N][K].
// 128x128 tile, BK=64, double-buffered LDS (1 barrier / K-step, 16 wmma / step)
#define BM 128
#define BN 128
#define BK 64
#define LDT 72  // padded LDS stride (elems): 144B rows -> conflict-free frags

__device__ __forceinline__ float gelu_exact(float x) {
  return 0.5f * x * (1.0f + erff(x * 0.70710678118654752f));
}

// Batch all global loads first (loads overlap), then cvt/store to LDS.
__device__ __forceinline__ void stage_gemm_tile(const float* __restrict__ A,
                                                const __bf16* __restrict__ Wt,
                                                __bf16* As, __bf16* Bs,
                                                int m0, int n0, int kt, int K,
                                                int tid) {
  float4 fa[8];
#pragma unroll
  for (int it = 0; it < 8; ++it) {
    const int e = (tid + it * 256) * 4;
    const int r = e >> 6, c = e & 63;
    fa[it] = *reinterpret_cast<const float4*>(A + (size_t)(m0 + r) * K + kt + c);
  }
  v8bf wb[4];
#pragma unroll
  for (int it = 0; it < 4; ++it) {
    const int g = tid + it * 256;
    const int n = g & 127, kc = (g >> 7) * 8;
    wb[it] = *reinterpret_cast<const v8bf*>(Wt + (size_t)(n0 + n) * K + kt + kc);
  }
#pragma unroll
  for (int it = 0; it < 8; ++it) {
    const int e = (tid + it * 256) * 4;
    const int r = e >> 6, c = e & 63;
    v4bf pk = { (__bf16)fa[it].x, (__bf16)fa[it].y, (__bf16)fa[it].z,
                (__bf16)fa[it].w };
    *reinterpret_cast<v4bf*>(&As[r * LDT + c]) = pk;
  }
#pragma unroll
  for (int it = 0; it < 4; ++it) {
    const int g = tid + it * 256;
    const int n = g & 127, kc = (g >> 7) * 8;
    *reinterpret_cast<v8bf*>(&Bs[n * LDT + kc]) = wb[it];
  }
}

__global__ __launch_bounds__(256) void gemm_bf16_kernel(
    const float* __restrict__ A, const __bf16* __restrict__ Wt,
    const float* __restrict__ bias, const float* __restrict__ res,
    float* __restrict__ out, int M, int N, int K, int act) {
  __shared__ __bf16 As[2][BM * LDT];
  __shared__ __bf16 Bs[2][BN * LDT];
  const int tid = threadIdx.x;
  const int lane = tid & 31, wave = tid >> 5;
  const int wm = wave & 3, wn = wave >> 2;   // 4x2 wave grid
  const int m0 = blockIdx.y * BM, n0 = blockIdx.x * BN;

  v8f acc[2][4];
#pragma unroll
  for (int i = 0; i < 2; ++i)
#pragma unroll
    for (int j = 0; j < 4; ++j) acc[i][j] = vzero8();

  stage_gemm_tile(A, Wt, As[0], Bs[0], m0, n0, 0, K, tid);
  __syncthreads();

  const int nk = K / BK;
  for (int t = 0; t < nk; ++t) {
    const int cur = t & 1;
    if (t + 1 < nk)  // overlap next tile's global loads with current WMMAs
      stage_gemm_tile(A, Wt, As[cur ^ 1], Bs[cur ^ 1], m0, n0, (t + 1) * BK, K, tid);

#pragma unroll
    for (int kk = 0; kk < BK; kk += 32) {
      v16bf af[2], bfr[4];
#pragma unroll
      for (int i = 0; i < 2; ++i)
        af[i] = frag_a(&As[cur][(wm * 32 + i * 16) * LDT + kk], LDT, lane);
#pragma unroll
      for (int j = 0; j < 4; ++j)
        bfr[j] = frag_b(&Bs[cur][(wn * 64 + j * 16) * LDT + kk], LDT, lane);
#pragma unroll
      for (int i = 0; i < 2; ++i)
#pragma unroll
        for (int j = 0; j < 4; ++j)
          acc[i][j] = wmma_bf16(af[i], bfr[j], acc[i][j]);
    }
    __syncthreads();
  }

  // ---- epilogue: bias + residual + activation ----
  const int half = lane >> 4, nn = lane & 15;
#pragma unroll
  for (int i = 0; i < 2; ++i)
#pragma unroll
    for (int j = 0; j < 4; ++j) {
      const int col = n0 + wn * 64 + j * 16 + nn;
      const float bv = bias[col];
#pragma unroll
      for (int r = 0; r < 8; ++r) {
        const size_t row = (size_t)(m0 + wm * 32 + i * 16 + half * 8 + r);
        float v = acc[i][j][r] + bv;
        if (res) v += res[row * N + col];
        if (act) v = gelu_exact(v);
        out[row * N + col] = v;
      }
    }
}

// ---------------- KV cache extraction ----------------
__global__ __launch_bounds__(256) void kvcopy_kernel(const float* __restrict__ qkv,
                                                     float* __restrict__ kd,
                                                     float* __restrict__ vd) {
  const size_t i = (size_t)blockIdx.x * 256 + threadIdx.x;  // B*S*E
  const size_t row = i / EE;
  const int e = (int)(i % EE);
  kd[i] = qkv[row * E3 + EE + e];
  vd[i] = qkv[row * E3 + 2 * EE + e];
}

// ---------------- flash-style causal attention ----------------
// grid: (S/64, B*H). block: 128 threads (4 waves); wave w owns 16 query rows.
#define ALD 72  // padded LDS stride for 64-wide tiles

__global__ __launch_bounds__(128) void attn_kernel(const float* __restrict__ qkv,
                                                   float* __restrict__ ctx) {
  __shared__ __bf16 Qs[64 * ALD];
  __shared__ __bf16 Ks[64 * ALD];
  __shared__ __bf16 Vt[64 * ALD];       // transposed: [d][kv]
  __shared__ __bf16 Ps[4][16 * ALD];    // per-wave P tile
  const int tid = threadIdx.x, lane = tid & 31, wave = tid >> 5;
  const int half = lane >> 4, nn = lane & 15;
  const int bh = blockIdx.y, b = bh / HH, h = bh % HH;
  const int q0 = blockIdx.x * 64;
  const size_t rowbase = (size_t)b * SS;
  const float scale = 0.125f;  // 1/sqrt(DH), folded into Q

  // stage Q (pre-scaled)
#pragma unroll
  for (int it = 0; it < 8; ++it) {
    const int e = (tid + it * 128) * 4;
    const int r = e >> 6, c = e & 63;
    const float4 f = *reinterpret_cast<const float4*>(
        qkv + (rowbase + q0 + r) * E3 + h * DH + c);
    v4bf pk = { (__bf16)(f.x * scale), (__bf16)(f.y * scale),
                (__bf16)(f.z * scale), (__bf16)(f.w * scale) };
    *reinterpret_cast<v4bf*>(&Qs[r * ALD + c]) = pk;
  }

  float mrow[8], lrow[8];
  v8f o[4];
#pragma unroll
  for (int r = 0; r < 8; ++r) { mrow[r] = -1.0e30f; lrow[r] = 0.0f; }
#pragma unroll
  for (int j = 0; j < 4; ++j) o[j] = vzero8();

  const int ktmax = blockIdx.x;  // kv tiles with kv0 <= q0
  for (int kt = 0; kt <= ktmax; ++kt) {
    const int kv0 = kt * 64;
    // stage K [kv][d] and V transposed [d][kv]
#pragma unroll
    for (int it = 0; it < 8; ++it) {
      const int e = (tid + it * 128) * 4;
      const int r = e >> 6, c = e & 63;
      const float4 fk = *reinterpret_cast<const float4*>(
          qkv + (rowbase + kv0 + r) * E3 + EE + h * DH + c);
      v4bf pk = { (__bf16)fk.x, (__bf16)fk.y, (__bf16)fk.z, (__bf16)fk.w };
      *reinterpret_cast<v4bf*>(&Ks[r * ALD + c]) = pk;
      const float4 fv = *reinterpret_cast<const float4*>(
          qkv + (rowbase + kv0 + r) * E3 + 2 * EE + h * DH + c);
      Vt[(c + 0) * ALD + r] = (__bf16)fv.x;
      Vt[(c + 1) * ALD + r] = (__bf16)fv.y;
      Vt[(c + 2) * ALD + r] = (__bf16)fv.z;
      Vt[(c + 3) * ALD + r] = (__bf16)fv.w;
    }
    __syncthreads();

    // scores: S = Q(16x64) @ K^T(64x64) for this wave's 16 rows
    v8f s[4];
#pragma unroll
    for (int j = 0; j < 4; ++j) s[j] = vzero8();
#pragma unroll
    for (int kk = 0; kk < 64; kk += 32) {
      const v16bf qa = frag_a(&Qs[(wave * 16) * ALD + kk], ALD, lane);
#pragma unroll
      for (int j = 0; j < 4; ++j)
        s[j] = wmma_bf16(qa, frag_b(&Ks[(j * 16) * ALD + kk], ALD, lane), s[j]);
    }

    // causal mask + additive tril bias (+1 on allowed) + online softmax
#pragma unroll
    for (int r = 0; r < 8; ++r) {
      const int qrow = q0 + wave * 16 + half * 8 + r;
      float mx = -1.0e30f;
#pragma unroll
      for (int j = 0; j < 4; ++j) {
        const int kvc = kv0 + j * 16 + nn;
        float sv = s[j][r];
        sv = (kvc <= qrow) ? (sv + 1.0f) : -1.0e9f;
        s[j][r] = sv;
        mx = fmaxf(mx, sv);
      }
#pragma unroll
      for (int d = 1; d < 16; d <<= 1) mx = fmaxf(mx, __shfl_xor(mx, d, 32));
      const float mnew = fmaxf(mrow[r], mx);
      const float alpha = __expf(mrow[r] - mnew);
      float psum = 0.0f;
#pragma unroll
      for (int j = 0; j < 4; ++j) {
        const float p = __expf(s[j][r] - mnew);
        s[j][r] = p;
        psum += p;
      }
#pragma unroll
      for (int d = 1; d < 16; d <<= 1) psum += __shfl_xor(psum, d, 32);
      lrow[r] = lrow[r] * alpha + psum;
      mrow[r] = mnew;
#pragma unroll
      for (int j = 0; j < 4; ++j) o[j][r] *= alpha;
    }

    // stage P (wave-private) and do O += P(16x64) @ V(64x64)
#pragma unroll
    for (int j = 0; j < 4; ++j)
#pragma unroll
      for (int r = 0; r < 8; ++r)
        Ps[wave][(half * 8 + r) * ALD + j * 16 + nn] = (__bf16)s[j][r];
#pragma unroll
    for (int kk = 0; kk < 64; kk += 32) {
      const v16bf pa = frag_a(&Ps[wave][kk], ALD, lane);
#pragma unroll
      for (int j = 0; j < 4; ++j)
        o[j] = wmma_bf16(pa, frag_b(&Vt[(j * 16) * ALD + kk], ALD, lane), o[j]);
    }
    __syncthreads();
  }

  // normalize + write ctx [B*S, E] with head offset
#pragma unroll
  for (int r = 0; r < 8; ++r) {
    const float inv = 1.0f / lrow[r];
    const size_t row = rowbase + (size_t)(q0 + wave * 16 + half * 8 + r);
#pragma unroll
    for (int j = 0; j < 4; ++j)
      ctx[row * EE + h * DH + j * 16 + nn] = o[j][r] * inv;
  }
}

// ---------------- last valid token gather ----------------
__global__ __launch_bounds__(256) void lastidx_kernel(const int* __restrict__ mask,
                                                      const float* __restrict__ hn,
                                                      float* __restrict__ lastst) {
  __shared__ int red[8];
  const int b = blockIdx.x;
  int best = -1;
  for (int s = threadIdx.x; s < SS; s += 256)
    if (mask[b * SS + s] != 0 && s > best) best = s;
  const int lane = threadIdx.x & 31, wave = threadIdx.x >> 5;
#pragma unroll
  for (int d = 16; d > 0; d >>= 1) best = max(best, __shfl_xor(best, d, 32));
  if (lane == 0) red[wave] = best;
  __syncthreads();
  int all = -1;
#pragma unroll
  for (int i = 0; i < 8; ++i) all = max(all, red[i]);
  if (all < 0) all = SS - 1;
  for (int e = threadIdx.x; e < EE; e += 256)
    lastst[b * EE + e] = hn[((size_t)b * SS + all) * EE + e];
}

// ---------------- LM head: logits = last_states @ wte^T ----------------
__global__ __launch_bounds__(256) void logits_kernel(const float* __restrict__ ls,
                                                     const float* __restrict__ wte,
                                                     float* __restrict__ logits) {
  __shared__ float lss[BB * EE];
  for (int i = threadIdx.x; i < BB * EE; i += 256) lss[i] = ls[i];
  __syncthreads();
  const int v = blockIdx.x * 256 + threadIdx.x;
  if (v >= VV) return;
  const float* wr = wte + (size_t)v * EE;
  float a0 = 0.f, a1 = 0.f, a2 = 0.f, a3 = 0.f;
  for (int e = 0; e < EE; e += 4) {
    const float4 w = *reinterpret_cast<const float4*>(wr + e);
    a0 += w.x * lss[e] + w.y * lss[e + 1] + w.z * lss[e + 2] + w.w * lss[e + 3];
    a1 += w.x * lss[EE + e] + w.y * lss[EE + e + 1] + w.z * lss[EE + e + 2] +
          w.w * lss[EE + e + 3];
    a2 += w.x * lss[2 * EE + e] + w.y * lss[2 * EE + e + 1] +
          w.z * lss[2 * EE + e + 2] + w.w * lss[2 * EE + e + 3];
    a3 += w.x * lss[3 * EE + e] + w.y * lss[3 * EE + e + 1] +
          w.z * lss[3 * EE + e + 2] + w.w * lss[3 * EE + e + 3];
  }
  logits[v] = a0;
  logits[VV + v] = a1;
  logits[2 * VV + v] = a2;
  logits[3 * VV + v] = a3;
}

// ---------------- vocab softmax ----------------
__global__ __launch_bounds__(256) void vsoftmax_kernel(const float* __restrict__ logits,
                                                       float* __restrict__ probs) {
  __shared__ float red[8];
  const int b = blockIdx.x;
  const float* lr = logits + (size_t)b * VV;
  float mx = -3.0e38f;
  for (int v = threadIdx.x; v < VV; v += 256) mx = fmaxf(mx, lr[v]);
  mx = block_max256(mx, red);
  float s = 0.0f;
  for (int v = threadIdx.x; v < VV; v += 256) s += __expf(lr[v] - mx);
  s = block_sum256(s, red);
  const float inv = 1.0f / s;
  for (int v = threadIdx.x; v < VV; v += 256)
    probs[(size_t)b * VV + v] = __expf(lr[v] - mx) * inv;
}

// ---------------- host orchestration ----------------
extern "C" void kernel_launch(void* const* d_in, const int* in_sizes, int n_in,
                              void* d_out, int out_size, void* d_ws, size_t ws_size,
                              hipStream_t stream) {
  (void)in_sizes; (void)n_in; (void)out_size; (void)ws_size;
  const int*   ids    = (const int*)d_in[0];
  const int*   amask  = (const int*)d_in[1];
  const float* wte    = (const float*)d_in[2];
  const float* wpe    = (const float*)d_in[3];
  const float* ln1w   = (const float*)d_in[4];
  const float* ln1b   = (const float*)d_in[5];
  const float* qkvw   = (const float*)d_in[6];
  const float* qkvb   = (const float*)d_in[7];
  const float* projw  = (const float*)d_in[8];
  const float* projb  = (const float*)d_in[9];
  // d_in[10] = attn_bias (tril, handled analytically in-kernel)
  const float* ln2w   = (const float*)d_in[11];
  const float* ln2b   = (const float*)d_in[12];
  const float* fcw    = (const float*)d_in[13];
  const float* fcb    = (const float*)d_in[14];
  const float* p2w    = (const float*)d_in[15];
  const float* p2b    = (const float*)d_in[16];
  const float* lnfw   = (const float*)d_in[17];
  const float* lnfb   = (const float*)d_in[18];

  float* probs = (float*)d_out;
  float* Kc = probs + (size_t)BB * VV;
  float* Vc = Kc + (size_t)LL * BB * SS * EE;

  // workspace carve
  char* ws = (char*)d_ws;
  size_t off = 0;
  auto carve = [&](size_t bytes) -> char* {
    char* p = ws + off;
    off += (bytes + 255) & ~(size_t)255;
    return p;
  };
  const size_t nQKVW = (size_t)LL * EE * E3;
  const size_t nPRJW = (size_t)LL * EE * EE;
  const size_t nFCW  = (size_t)LL * EE * FF;
  __bf16* wqkv_bf = (__bf16*)carve(nQKVW * 2);  // [L][N=3E][K=E]  (transposed)
  __bf16* wprj_bf = (__bf16*)carve(nPRJW * 2);  // [L][N=E][K=E]
  __bf16* wfc_bf  = (__bf16*)carve(nFCW * 2);   // [L][N=F][K=E]
  __bf16* wp2_bf  = (__bf16*)carve(nFCW * 2);   // [L][N=E][K=F]
  float* x      = (float*)carve((size_t)BB * SS * EE * 4);
  float* hn     = (float*)carve((size_t)BB * SS * EE * 4);
  float* qkvbuf = (float*)carve((size_t)BB * SS * E3 * 4);
  float* ctxbuf = (float*)carve((size_t)BB * SS * EE * 4);
  float* fcact  = (float*)carve((size_t)BB * SS * FF * 4);
  float* lastst = (float*)carve((size_t)BB * EE * 4);
  float* logitb = (float*)carve((size_t)BB * VV * 4);

  // transpose+convert weights to bf16 [N][K] (one streaming pass, L2-buffered)
  cvtT_bf16_kernel<<<dim3((unsigned)((size_t)EE * E3 / 1024), LL), 256, 0, stream>>>(
      qkvw, wqkv_bf, EE, E3);
  cvtT_bf16_kernel<<<dim3((unsigned)((size_t)EE * EE / 1024), LL), 256, 0, stream>>>(
      projw, wprj_bf, EE, EE);
  cvtT_bf16_kernel<<<dim3((unsigned)((size_t)EE * FF / 1024), LL), 256, 0, stream>>>(
      fcw, wfc_bf, EE, FF);
  cvtT_bf16_kernel<<<dim3((unsigned)((size_t)FF * EE / 1024), LL), 256, 0, stream>>>(
      p2w, wp2_bf, FF, EE);

  // embedding
  embed_kernel<<<BB * SS, 192, 0, stream>>>(ids, wte, wpe, x);

  const int M = BB * SS;  // 4096
  for (int l = 0; l < LL; ++l) {
    // ln1 -> hn ; qkv = hn @ Wqkv + b
    ln_kernel<<<M, 256, 0, stream>>>(x, ln1w + l * EE, ln1b + l * EE, hn);
    gemm_bf16_kernel<<<dim3(E3 / BN, M / BM), 256, 0, stream>>>(
        hn, wqkv_bf + (size_t)l * EE * E3, qkvb + l * E3, nullptr, qkvbuf,
        M, E3, EE, 0);
    // KV cache slices -> d_out
    kvcopy_kernel<<<(M * EE) / 256, 256, 0, stream>>>(
        qkvbuf, Kc + (size_t)l * M * EE, Vc + (size_t)l * M * EE);
    // attention -> ctx
    attn_kernel<<<dim3(SS / 64, BB * HH), 128, 0, stream>>>(qkvbuf, ctxbuf);
    // x += ctx @ Wproj + b
    gemm_bf16_kernel<<<dim3(EE / BN, M / BM), 256, 0, stream>>>(
        ctxbuf, wprj_bf + (size_t)l * EE * EE, projb + l * EE, x, x,
        M, EE, EE, 0);
    // ln2 -> hn ; m = gelu(hn @ Wfc + b)
    ln_kernel<<<M, 256, 0, stream>>>(x, ln2w + l * EE, ln2b + l * EE, hn);
    gemm_bf16_kernel<<<dim3(FF / BN, M / BM), 256, 0, stream>>>(
        hn, wfc_bf + (size_t)l * EE * FF, fcb + l * FF, nullptr, fcact,
        M, FF, EE, 1);
    // x += m @ Wproj2 + b
    gemm_bf16_kernel<<<dim3(EE / BN, M / BM), 256, 0, stream>>>(
        fcact, wp2_bf + (size_t)l * FF * EE, p2b + l * EE, x, x,
        M, EE, FF, 0);
  }

  // final LN, last-token gather, LM head, softmax
  ln_kernel<<<M, 256, 0, stream>>>(x, lnfw, lnfb, hn);
  lastidx_kernel<<<BB, 256, 0, stream>>>(amask, hn, lastst);
  logits_kernel<<<(VV + 255) / 256, 256, 0, stream>>>(lastst, wte, logitb);
  vsoftmax_kernel<<<BB, 256, 0, stream>>>(logitb, probs);
}
